// EdgeDecoder_10720238371045
// MI455X (gfx1250) — matrix operand order repaired and checked by reading, not compiled, not run on hardware
//
#include <hip/hip_runtime.h>

typedef __attribute__((ext_vector_type(16))) _Float16 v16h;
typedef __attribute__((ext_vector_type(8)))  float    v8f;

#define WAVES_PER_BLOCK 8
#define THREADS (WAVES_PER_BLOCK * 32)

// ---------------------------------------------------------------------------
// Prep: split W1 (f32 [128,128], row-major [k][n]) into hi/lo f16, stored in
// the B-fragment layout this kernel uses with v_wmma_f32_16x16x32_f16 (k order matches A):
//   fragment index t = (kt*8 + nt)*32 + lane,  each holding 16 halfs:
//   lane -> column n = nt*16 + (lane&15); K base = kt*32 + (lane>>4)*16
//   half i -> W1[Kbase + i][n]
// ---------------------------------------------------------------------------
__global__ void prep_w1_split(const float* __restrict__ W1,
                              _Float16* __restrict__ w1hi,
                              _Float16* __restrict__ w1lo) {
    int t = blockIdx.x * blockDim.x + threadIdx.x;
    if (t >= 4 * 8 * 32) return;            // kt(4) x nt(8) x lane(32)
    int lane = t & 31;
    int nt   = (t >> 5) & 7;
    int kt   = t >> 8;
    int n    = nt * 16 + (lane & 15);
    int kb   = kt * 32 + (lane >> 4) * 16;
    size_t base = (size_t)t * 16;
#pragma unroll
    for (int i = 0; i < 16; ++i) {
        float w = W1[(size_t)(kb + i) * 128 + n];
        _Float16 hi = (_Float16)w;
        _Float16 lo = (_Float16)(w - (float)hi);
        w1hi[base + i] = hi;
        w1lo[base + i] = lo;
    }
}

// ---------------------------------------------------------------------------
// Main: each wave32 computes 16 edges:
//   z[16,128] = z_src[row]*z_dst[col] (split hi/lo f16)
//   C[16,128] = z @ W1  via WMMA (3-product hi/lo compensation, f32 acc)
//   out[e]    = relu(C + b1) . W2 + b2
// ---------------------------------------------------------------------------
__global__ void __launch_bounds__(THREADS)
edge_mlp_wmma(const float* __restrict__ z_src,
              const float* __restrict__ z_dst,
              const int*   __restrict__ row,
              const int*   __restrict__ col,
              const float* __restrict__ b1,
              const float* __restrict__ W2,
              const float* __restrict__ b2,
              const _Float16* __restrict__ w1hi,
              const _Float16* __restrict__ w1lo,
              float* __restrict__ out,
              int E, int nTiles) {
    int wave = blockIdx.x * WAVES_PER_BLOCK + ((int)threadIdx.x >> 5);
    if (wave >= nTiles) return;             // wave-uniform exit (EXEC stays full)

    int lane  = (int)threadIdx.x & 31;
    int l16   = lane & 15;
    int khalf = lane >> 4;

    // Edge handled by this lane's A-fragment row (clamped for a partial tail)
    int e = wave * 16 + l16;
    if (e >= E) e = E - 1;
    int r = row[e];
    int c = col[e];
    const float* zs = z_src + (size_t)r * 128;
    const float* zd = z_dst + (size_t)c * 128;

    v8f acc[8] = {};                        // C tiles for N = nt*16..+15

#pragma unroll
    for (int kt = 0; kt < 4; ++kt) {
        // Build hi/lo A fragments: halfs [g*8+i] <-> K = kt*32 + khalf*16 + g*8 + i
        v16h ah = {}, al = {};
#pragma unroll
        for (int g = 0; g < 2; ++g) {
            int kb = kt * 32 + khalf * 16 + g * 8;     // 32B-aligned
            float4 sa = ((const float4*)(zs + kb))[0];
            float4 sb = ((const float4*)(zs + kb))[1];
            float4 da = ((const float4*)(zd + kb))[0];
            float4 db = ((const float4*)(zd + kb))[1];
            float sv[8] = {sa.x, sa.y, sa.z, sa.w, sb.x, sb.y, sb.z, sb.w};
            float dv[8] = {da.x, da.y, da.z, da.w, db.x, db.y, db.z, db.w};
#pragma unroll
            for (int i = 0; i < 8; ++i) {
                float z = sv[i] * dv[i];
                _Float16 h = (_Float16)z;
                ah[g * 8 + i] = h;
                al[g * 8 + i] = (_Float16)(z - (float)h);
            }
        }
        // 8 N-tiles against prelaid B fragments of W1 (hi/lo)
#pragma unroll
        for (int nt = 0; nt < 8; ++nt) {
            size_t off = ((size_t)((kt * 8 + nt) * 32 + lane)) * 16;
            v16h bh = *(const v16h*)(w1hi + off);
            v16h bl = *(const v16h*)(w1lo + off);
            acc[nt] = __builtin_amdgcn_wmma_f32_16x16x32_f16(
                false, al, false, bh, (short)0, acc[nt], false, false);
            acc[nt] = __builtin_amdgcn_wmma_f32_16x16x32_f16(
                false, ah, false, bl, (short)0, acc[nt], false, false);
            acc[nt] = __builtin_amdgcn_wmma_f32_16x16x32_f16(
                false, ah, false, bh, (short)0, acc[nt], false, false);
        }
    }

    // Epilogue: C layout -> lane holds column n = nt*16+l16, rows M = khalf*8+j
    float partial[8] = {0.f, 0.f, 0.f, 0.f, 0.f, 0.f, 0.f, 0.f};
#pragma unroll
    for (int nt = 0; nt < 8; ++nt) {
        int n = nt * 16 + l16;
        float bias = b1[n];
        float w2n  = W2[n];
#pragma unroll
        for (int j = 0; j < 8; ++j) {
            float h = acc[nt][j] + bias;
            h = fmaxf(h, 0.0f);
            partial[j] = fmaf(h, w2n, partial[j]);
        }
    }
    // Butterfly reduce across each 16-lane half (xor masks stay in-group)
#pragma unroll
    for (int j = 0; j < 8; ++j) {
        partial[j] += __shfl_xor(partial[j], 1, 32);
        partial[j] += __shfl_xor(partial[j], 2, 32);
        partial[j] += __shfl_xor(partial[j], 4, 32);
        partial[j] += __shfl_xor(partial[j], 8, 32);
    }
    // lane l16 (<8) emits row khalf*8 + l16  -> coalesced 16-wide store
    float val = partial[0];
#pragma unroll
    for (int j = 1; j < 8; ++j)
        if (l16 == j) val = partial[j];
    int m = wave * 16 + khalf * 8 + l16;
    if (l16 < 8 && m < E) out[m] = val + b2[0];
}

extern "C" void kernel_launch(void* const* d_in, const int* in_sizes, int n_in,
                              void* d_out, int out_size, void* d_ws, size_t ws_size,
                              hipStream_t stream) {
    const float* z_src = (const float*)d_in[0];
    const float* z_dst = (const float*)d_in[1];
    const int*   row   = (const int*)d_in[2];
    const int*   col   = (const int*)d_in[3];
    const float* W1    = (const float*)d_in[4];
    const float* b1    = (const float*)d_in[5];
    const float* W2    = (const float*)d_in[6];
    const float* b2    = (const float*)d_in[7];
    float*       out   = (float*)d_out;

    int E = in_sizes[2];                     // number of edges

    _Float16* w1hi = (_Float16*)d_ws;        // 4*8*32*16 halfs = 32 KB
    _Float16* w1lo = w1hi + 4 * 8 * 32 * 16; // + 32 KB (ws >= 64 KB)

    prep_w1_split<<<4, 256, 0, stream>>>(W1, w1hi, w1lo);

    int nTiles = (E + 15) / 16;
    int blocks = (nTiles + WAVES_PER_BLOCK - 1) / WAVES_PER_BLOCK;
    edge_mlp_wmma<<<blocks, THREADS, 0, stream>>>(
        z_src, z_dst, row, col, b1, W2, b2, w1hi, w1lo, out, E, nTiles);
}
